// DistributedTriangleMultiplicationCpp_62182536511498
// MI455X (gfx1250) — matrix operand order repaired and checked by reading, not compiled, not run on hardware
//
#include <hip/hip_runtime.h>
#include <hip/hip_bf16.h>

#define N_TOK 768
#define CCH   128
#define LNEPS 1e-5f

typedef __attribute__((ext_vector_type(16))) __bf16        v16bf;
typedef __attribute__((ext_vector_type(8)))  float         v8f;
typedef __attribute__((ext_vector_type(4)))  unsigned int  v4u;

union ABPack  { v4u u[2]; v16bf v; };
union Pack8bf { __hip_bfloat16 h[8]; v4u u; };

// Fast sigmoid: v_exp_f32 + v_rcp_f32 (no precise-division fixup chain).
__device__ __forceinline__ float sigmoidf_(float x) {
    return __builtin_amdgcn_rcpf(1.0f + __expf(-x));
}

__device__ __forceinline__ v8f wmma_bf16(const ABPack& A, const ABPack& B, v8f acc) {
    return __builtin_amdgcn_wmma_f32_16x16x32_bf16(
        false, A.v, false, B.v, (short)0, acc, false, false);
}

// ---------------------------------------------------------------------------
// Kernel 0: convert all weights to bf16, concatenated [768][128]:
//   rows   0..255 : w_proj     256..511 : w_gate
//   rows 512..639 : w_gl       640..767 : w_out
// ---------------------------------------------------------------------------
__global__ void cvt_weights(const float* __restrict__ w_proj,
                            const float* __restrict__ w_gate,
                            const float* __restrict__ w_gl,
                            const float* __restrict__ w_out,
                            __hip_bfloat16* __restrict__ wcat)
{
    const int idx = blockIdx.x * 256 + threadIdx.x;   // 0 .. 768*128-1
    const int row = idx >> 7;
    float v;
    if      (row < 256) v = w_proj[idx];
    else if (row < 512) v = w_gate[idx - 256 * CCH];
    else if (row < 640) v = w_gl[idx - 512 * CCH];
    else                v = w_out[idx - 640 * CCH];
    wcat[idx] = __float2bfloat16(v);
}

// ---------------------------------------------------------------------------
// LayerNorm over C=128, one 128-thread block per token; bf16 output [tok][c].
// Used for both LN1 (act) and LN2 (tri).
// ---------------------------------------------------------------------------
__global__ void ln_kernel(const float* __restrict__ x_in,
                          const float* __restrict__ gw,
                          const float* __restrict__ gb,
                          __hip_bfloat16* __restrict__ xout)
{
    const int t = threadIdx.x;
    const size_t tok = (size_t)blockIdx.x * N_TOK + blockIdx.y;
    __shared__ float s1[CCH], s2[CCH];
    const float v = x_in[tok * CCH + t];
    s1[t] = v; s2[t] = v * v;
    __syncthreads();
    for (int s = CCH / 2; s > 0; s >>= 1) {
        if (t < s) { s1[t] += s1[t + s]; s2[t] += s2[t + s]; }
        __syncthreads();
    }
    const float mu  = s1[0] * (1.0f / CCH);
    const float var = s2[0] * (1.0f / CCH) - mu * mu;
    const float inv = __builtin_amdgcn_rsqf(var + LNEPS);   // fast v_rsq_f32
    xout[tok * CCH + t] = __float2bfloat16((v - mu) * inv * gw[t] + gb[t]);
}

// ---------------------------------------------------------------------------
// Kernel 1b: WMMA projections. M = 589824 tokens, K = 128.
//   blockIdx.y < 16 : paired tiles (proj[d], gate[d]) -> a/b (mask*sigmoid fused)
//                     packed 16-B bf16 stores into channel-major [c][i][j]
//   blockIdx.y >= 16: gating_linear tiles -> g2 = sigmoid(x @ w_gl^T), [tok][c]
// Each wave: 16 tokens x 16 outputs. 8 waves/block share the same weight tile.
// ---------------------------------------------------------------------------
__global__ void __launch_bounds__(256)
proj_gemm(const unsigned short* __restrict__ xn,    // [tok][128] bf16
          const unsigned short* __restrict__ wcat,  // [768][128] bf16
          const float* __restrict__ mask,
          __hip_bfloat16* __restrict__ aT,
          __hip_bfloat16* __restrict__ bT,
          __hip_bfloat16* __restrict__ g2)
{
    const int lane = threadIdx.x & 31;
    const int wave = threadIdx.x >> 5;
    const int hi   = lane >> 4;
    const int rm   = lane & 15;
    const int t0   = (blockIdx.x * 8 + wave) * 16;
    const int y    = blockIdx.y;
    const int aK   = hi ? 16 : 0;   // A 16x32: v0-3 @ aK, v4-7 @ aK+8
    const int bK   = hi ? 16 : 0;   // B 32x16: v0-3 @ bK, v4-7 @ bK+8
    const int mbase = hi ? 8 : 0;

    const unsigned short* __restrict__ aRow = xn + (size_t)(t0 + rm) * CCH;

    if (y < 16) {
        const int d0 = y * 16;
        const unsigned short* __restrict__ bpRow = wcat + (size_t)(d0 + rm) * CCH;
        const unsigned short* __restrict__ bgRow = wcat + (size_t)(256 + d0 + rm) * CCH;
        v8f accp = {}, accg = {};
#pragma unroll
        for (int k0 = 0; k0 < CCH; k0 += 32) {
            ABPack A, Bp, Bg;
            A.u[0]  = *(const v4u*)(aRow + k0 + aK);
            A.u[1]  = *(const v4u*)(aRow + k0 + aK + 8);
            Bp.u[0] = *(const v4u*)(bpRow + k0 + bK);
            Bp.u[1] = *(const v4u*)(bpRow + k0 + bK + 8);
            Bg.u[0] = *(const v4u*)(bgRow + k0 + bK);
            Bg.u[1] = *(const v4u*)(bgRow + k0 + bK + 8);
            accp = wmma_bf16(A, Bp, accp);
            accg = wmma_bf16(A, Bg, accg);
        }
        const int tb = t0 + mbase;                 // 8 consecutive tokens (same i)
        Pack8bf res;
#pragma unroll
        for (int r = 0; r < 8; ++r) {
            const float mk = mask[tb + r];
            res.h[r] = __float2bfloat16(accp[r] * mk * sigmoidf_(accg[r]));
        }
        const int i = tb / N_TOK;
        const int j = tb % N_TOK;
        __hip_bfloat16* __restrict__ dst = (d0 < CCH) ? aT : bT;
        const int dch = ((d0 < CCH) ? d0 : d0 - CCH) + rm;
        dst += (size_t)dch * N_TOK * N_TOK + (size_t)i * N_TOK + j;
        *(v4u*)dst = res.u;                         // 8 bf16, contiguous in j
    } else {
        const int g = y - 16;                       // gl tile 0..7
        const unsigned short* __restrict__ bRow =
            wcat + (size_t)(512 + g * 16 + rm) * CCH;
        v8f acc = {};
#pragma unroll
        for (int k0 = 0; k0 < CCH; k0 += 32) {
            ABPack A, B;
            A.u[0] = *(const v4u*)(aRow + k0 + aK);
            A.u[1] = *(const v4u*)(aRow + k0 + aK + 8);
            B.u[0] = *(const v4u*)(bRow + k0 + bK);
            B.u[1] = *(const v4u*)(bRow + k0 + bK + 8);
            acc = wmma_bf16(A, B, acc);
        }
        const int dch = g * 16 + rm;
#pragma unroll
        for (int r = 0; r < 8; ++r) {
            g2[(size_t)(t0 + mbase + r) * CCH + dch] =
                __float2bfloat16(sigmoidf_(acc[r]));
        }
    }
}

// ---------------------------------------------------------------------------
// Kernel 2: triangle contraction out[i,j,c] = sum_k a[i,k,c]*b[j,k,c].
// 2x2 register tiling: each wave computes a 32x32 (i,j) tile for one channel,
// reusing each A/B fragment twice (8 b128 loads -> 4 WMMAs per k-chunk).
// grid.z groups 8 channels/block for L2 locality (~2.25 MB/channel << 192 MB).
// ---------------------------------------------------------------------------
__global__ void __launch_bounds__(256)
tri_gemm(const unsigned short* __restrict__ aT,
         const unsigned short* __restrict__ bT,
         float* __restrict__ tri)
{
    const int lane = threadIdx.x & 31;
    const int wave = threadIdx.x >> 5;
    const int i0   = blockIdx.x * 32;
    const int j0   = blockIdx.y * 32;
    const int c    = blockIdx.z * 8 + wave;
    const int hi   = lane >> 4;
    const int rm   = lane & 15;

    const size_t plane = (size_t)c * N_TOK * N_TOK;
    const unsigned short* __restrict__ aRow0 = aT + plane + (size_t)(i0 + rm) * N_TOK;
    const unsigned short* __restrict__ aRow1 = aRow0 + (size_t)16 * N_TOK;
    const unsigned short* __restrict__ bRow0 = bT + plane + (size_t)(j0 + rm) * N_TOK;
    const unsigned short* __restrict__ bRow1 = bRow0 + (size_t)16 * N_TOK;
    const int aK = hi ? 16 : 0;
    const int bK = hi ? 16 : 0;

    v8f acc00 = {}, acc01 = {}, acc10 = {}, acc11 = {};
    for (int k0 = 0; k0 < N_TOK; k0 += 32) {
        ABPack A0, A1, B0, B1;
        A0.u[0] = *(const v4u*)(aRow0 + k0 + aK);
        A0.u[1] = *(const v4u*)(aRow0 + k0 + aK + 8);
        A1.u[0] = *(const v4u*)(aRow1 + k0 + aK);
        A1.u[1] = *(const v4u*)(aRow1 + k0 + aK + 8);
        B0.u[0] = *(const v4u*)(bRow0 + k0 + bK);
        B0.u[1] = *(const v4u*)(bRow0 + k0 + bK + 8);
        B1.u[0] = *(const v4u*)(bRow1 + k0 + bK);
        B1.u[1] = *(const v4u*)(bRow1 + k0 + bK + 8);
        __builtin_prefetch((const void*)(aRow0 + k0 + 64), 0, 0);
        __builtin_prefetch((const void*)(aRow1 + k0 + 64), 0, 0);
        __builtin_prefetch((const void*)(bRow0 + k0 + 64), 0, 0);
        __builtin_prefetch((const void*)(bRow1 + k0 + 64), 0, 0);
        acc00 = wmma_bf16(A0, B0, acc00);
        acc01 = wmma_bf16(A0, B1, acc01);
        acc10 = wmma_bf16(A1, B0, acc10);
        acc11 = wmma_bf16(A1, B1, acc11);
    }

    const int mbase = hi ? 8 : 0;
#pragma unroll
    for (int r = 0; r < 8; ++r) {
        const size_t ri0 = (size_t)(i0 + mbase + r) * N_TOK;
        const size_t ri1 = (size_t)(i0 + 16 + mbase + r) * N_TOK;
        tri[(ri0 + (j0 + rm))      * CCH + c] = acc00[r];
        tri[(ri0 + (j0 + 16 + rm)) * CCH + c] = acc01[r];
        tri[(ri1 + (j0 + rm))      * CCH + c] = acc10[r];
        tri[(ri1 + (j0 + 16 + rm)) * CCH + c] = acc11[r];
    }
}

// ---------------------------------------------------------------------------
// Kernel 3b: WMMA output projection: out = (z @ w_out^T) * g2.
// Each wave: 16 tokens x 16 outputs, K=128.
// ---------------------------------------------------------------------------
__global__ void __launch_bounds__(256)
out_gemm(const unsigned short* __restrict__ z,     // [tok][128] bf16 (LN2'd)
         const unsigned short* __restrict__ wcat,  // rows 640..767 = w_out
         const __hip_bfloat16* __restrict__ g2,
         float* __restrict__ out)
{
    const int lane = threadIdx.x & 31;
    const int wave = threadIdx.x >> 5;
    const int hi   = lane >> 4;
    const int rm   = lane & 15;
    const int t0   = (blockIdx.x * 8 + wave) * 16;
    const int d0   = blockIdx.y * 16;
    const int aK   = hi ? 16 : 0;
    const int bK   = hi ? 16 : 0;

    const unsigned short* __restrict__ aRow = z + (size_t)(t0 + rm) * CCH;
    const unsigned short* __restrict__ bRow = wcat + (size_t)(640 + d0 + rm) * CCH;

    v8f acc = {};
#pragma unroll
    for (int k0 = 0; k0 < CCH; k0 += 32) {
        ABPack A, B;
        A.u[0] = *(const v4u*)(aRow + k0 + aK);
        A.u[1] = *(const v4u*)(aRow + k0 + aK + 8);
        B.u[0] = *(const v4u*)(bRow + k0 + bK);
        B.u[1] = *(const v4u*)(bRow + k0 + bK + 8);
        acc = wmma_bf16(A, B, acc);
    }

    const int mbase = hi ? 8 : 0;
    const int d = d0 + rm;
#pragma unroll
    for (int r = 0; r < 8; ++r) {
        const size_t t = (size_t)(t0 + mbase + r);
        out[t * CCH + d] = acc[r] * __bfloat162float(g2[t * CCH + d]);
    }
}

// ---------------------------------------------------------------------------
extern "C" void kernel_launch(void* const* d_in, const int* in_sizes, int n_in,
                              void* d_out, int out_size, void* d_ws, size_t ws_size,
                              hipStream_t stream)
{
    (void)in_sizes; (void)n_in; (void)out_size; (void)ws_size;

    const float* act    = (const float*)d_in[0];
    const float* mask   = (const float*)d_in[1];
    const float* ln1_w  = (const float*)d_in[2];
    const float* ln1_b  = (const float*)d_in[3];
    const float* w_proj = (const float*)d_in[4];
    const float* w_gate = (const float*)d_in[5];
    const float* ln2_w  = (const float*)d_in[6];
    const float* ln2_b  = (const float*)d_in[7];
    const float* w_out  = (const float*)d_in[8];
    const float* w_gl   = (const float*)d_in[9];
    float* out = (float*)d_out;

    // Workspace carve-up; plane = 128*768*768 = 75,497,472 elements.
    //   xn/z (aliased) 151MB | aT 151MB | bT 151MB | g2 151MB | tri 302MB | wcat 192KB
    const size_t plane = (size_t)CCH * N_TOK * N_TOK;
    char* ws = (char*)d_ws;
    __hip_bfloat16* xn   = (__hip_bfloat16*)(ws);                 // LN1 out; reused as z
    __hip_bfloat16* aT   = (__hip_bfloat16*)(ws + plane * 2);
    __hip_bfloat16* bT   = (__hip_bfloat16*)(ws + plane * 4);
    __hip_bfloat16* g2   = (__hip_bfloat16*)(ws + plane * 6);
    float*          tri  = (float*)(ws + plane * 8);
    __hip_bfloat16* wcat = (__hip_bfloat16*)(ws + plane * 8 + plane * 4);

    cvt_weights<<<(768 * CCH) / 256, 256, 0, stream>>>(w_proj, w_gate, w_gl, w_out, wcat);

    dim3 gtok(N_TOK, N_TOK);
    ln_kernel<<<gtok, CCH, 0, stream>>>(act, ln1_w, ln1_b, xn);

    dim3 gproj((N_TOK * N_TOK) / (8 * 16), 24);
    proj_gemm<<<gproj, 256, 0, stream>>>((const unsigned short*)xn,
                                         (const unsigned short*)wcat,
                                         mask, aT, bT, g2);

    dim3 ggemm(N_TOK / 32, N_TOK / 32, CCH / 8);
    tri_gemm<<<ggemm, 256, 0, stream>>>((const unsigned short*)aT,
                                        (const unsigned short*)bT, tri);

    ln_kernel<<<gtok, CCH, 0, stream>>>(tri, ln2_w, ln2_b, xn);   // z aliases xn

    dim3 gout((N_TOK * N_TOK) / (8 * 16), CCH / 16);
    out_gemm<<<gout, 256, 0, stream>>>((const unsigned short*)xn,
                                       (const unsigned short*)wcat, g2, out);
}
